// COGV1_41274635714985
// MI455X (gfx1250) — matrix operand order repaired and checked by reading, not compiled, not run on hardware
//
#include <hip/hip_runtime.h>

#define NB   32
#define OCH  128
#define ICH  3
#define IMG  224
#define O1   112
#define PO   56
#define POP  58      // padded (halo) pooled plane
#define KTOT 1152    // 128*9
#define PIXT 100352  // 32*56*56

typedef __attribute__((ext_vector_type(16))) _Float16 v16h;
typedef __attribute__((ext_vector_type(8)))  float    v8f;
typedef __attribute__((ext_vector_type(4)))  float    v4f;

union Frag16 { v16h h; v4f f[2]; };

// ---------------------------------------------------------------------------
// Window geometry: replicates the numpy _windows() math on-device.
// borders = linspace(0,112,8) = multiples of 16 (exact).
// ---------------------------------------------------------------------------
__device__ __forceinline__ void window_params(int s, int w,
    int& ti, int& li, int& bi, int& ri,
    int& to, int& lo, int& bo, int& ro,
    int& fh, int& fw)
{
    int a = s * 16, b2 = (s + 1) * 16, c = IMG - b2, d = IMG - a;
    int q0, q1, q2, q3; // (top,left,bottom,right) in padded-image coords
    if      (w == 0) { q0 = a;  q1 = a;  q2 = b2; q3 = c;  }
    else if (w == 1) { q0 = b2; q1 = a;  q2 = d;  q3 = b2; }
    else if (w == 2) { q0 = c;  q1 = b2; q2 = d;  q3 = d;  }
    else             { q0 = a;  q1 = c;  q2 = c;  q3 = d;  }
    to = q0 >> 1; lo = q1 >> 1; bo = q2 >> 1; ro = q3 >> 1;
    float sc = (float)(2.0 - (double)s * (1.0 / 6.0)); // linspace(2,1,7)[s]
    int p0 = (int)((float)(q0 + 6) / sc);
    int p1 = (int)((float)(q1 + 6) / sc);
    int p2 = (int)((float)(q2 + 6) / sc);
    int p3 = (int)((float)(q3 + 6) / sc);
    ti = p0 - 3; li = p1 - 3; bi = p2 + 3; ri = p3 + 3;
    fh = (bi - ti - 7) / 2 + 1;
    fw = (ri - li - 7) / 2 + 1;
}

// ---------------------------------------------------------------------------
// Kernel 1: multi-window conv1 (7x7 s2) + nearest upsample scatter -> out1.
// block: 128 threads (1 per output channel). Each block handles an 8-pixel
// strip of the 112x112 plane for one batch image. Weights (18816 f32) live in
// dynamic LDS; the 147 window taps per pixel are staged in LDS too.
// ---------------------------------------------------------------------------
__global__ void conv1_kernel(const float* __restrict__ inp,
                             const float* __restrict__ w1,
                             float* __restrict__ out1)
{
    extern __shared__ float smem[];
    float* sW  = smem;             // 128*147
    float* sIn = smem + 128 * 147; // 147
    const int tid = threadIdx.x;
    const int n = blockIdx.y;
    const int strip = blockIdx.x;

    for (int i = tid; i < 128 * 147; i += 128) sW[i] = w1[i];
    __syncthreads();

    for (int j = 0; j < 8; ++j) {
        int pix = strip * 8 + j;
        int oy = pix / O1, ox = pix % O1;

        // last (s,w) window covering this pixel wins (replicates .set order)
        int ti = 0, li = 0, bi = 7, ri = 7, to = 0, lo = 0, bo = 1, ro = 1;
        int fh = 1, fw = 1; int found = 0;
        for (int s = 0; s < 7; ++s)
            for (int w = 0; w < 4; ++w) {
                int a0,a1,a2,a3,b0,b1,b2,b3,c0,c1;
                window_params(s, w, a0,a1,a2,a3, b0,b1,b2,b3, c0,c1);
                if (oy >= b0 && oy < b2 && ox >= b1 && ox < b3) {
                    ti=a0; li=a1; bi=a2; ri=a3; to=b0; lo=b1; bo=b2; ro=b3;
                    fh=c0; fw=c1; found=1;
                }
            }
        int fy = ((oy - to) * fh) / (bo - to);
        int fx = ((ox - lo) * fw) / (ro - lo);

        __syncthreads();
        for (int idx = tid; idx < 147; idx += 128) {
            int c = idx / 49, r = idx % 49, ky = r / 7, kx = r % 7;
            int ih = ti + fy * 2 + ky - 3;   // padded -> input coords
            int iw = li + fx * 2 + kx - 3;
            float v = 0.f;
            if (found && ih >= 0 && ih < IMG && iw >= 0 && iw < IMG)
                v = inp[((n * ICH + c) * IMG + ih) * IMG + iw];
            sIn[idx] = v;
        }
        __syncthreads();

        float sum = 0.f;
        #pragma unroll 7
        for (int idx = 0; idx < 147; ++idx)
            sum = fmaf(sIn[idx], sW[tid * 147 + idx], sum);
        out1[((size_t)(n * OCH + tid) * O1 + oy) * O1 + ox] = found ? sum : 0.f;
    }
}

// ---------------------------------------------------------------------------
// Kernel 2: per-channel BN stats over NCHW f32 buffer -> scale/shift (256 f)
// ---------------------------------------------------------------------------
__global__ void bn_stats_nchw(const float* __restrict__ x,
                              const float* __restrict__ gamma,
                              const float* __restrict__ beta,
                              float* __restrict__ ss, int HW, int N)
{
    const int c = blockIdx.x, tid = threadIdx.x;
    const long total = (long)N * HW;
    float s = 0.f, s2 = 0.f;
    for (long i = tid; i < total; i += blockDim.x) {
        long nn = i / HW, sp = i % HW;
        float v = x[((nn * OCH + c) * (long)HW) + sp];
        s += v; s2 += v * v;
    }
    __shared__ float r1[256], r2[256];
    r1[tid] = s; r2[tid] = s2; __syncthreads();
    for (int off = 128; off > 0; off >>= 1) {
        if (tid < off) { r1[tid] += r1[tid + off]; r2[tid] += r2[tid + off]; }
        __syncthreads();
    }
    if (tid == 0) {
        float mean = r1[0] / (float)total;
        float var  = r2[0] / (float)total - mean * mean;
        float sc = gamma[c] * rsqrtf(var + 1e-5f);
        ss[c] = sc; ss[OCH + c] = beta[c] - mean * sc;
    }
}

// Same, for the pixel-major conv2 output ([pixel][oc])
__global__ void bn_stats_pix(const float* __restrict__ x,
                             const float* __restrict__ gamma,
                             const float* __restrict__ beta,
                             float* __restrict__ ss, int P)
{
    const int c = blockIdx.x, tid = threadIdx.x;
    float s = 0.f, s2 = 0.f;
    for (int p = tid; p < P; p += blockDim.x) {
        float v = x[(size_t)p * OCH + c];
        s += v; s2 += v * v;
    }
    __shared__ float r1[256], r2[256];
    r1[tid] = s; r2[tid] = s2; __syncthreads();
    for (int off = 128; off > 0; off >>= 1) {
        if (tid < off) { r1[tid] += r1[tid + off]; r2[tid] += r2[tid + off]; }
        __syncthreads();
    }
    if (tid == 0) {
        float mean = r1[0] / (float)P;
        float var  = r2[0] / (float)P - mean * mean;
        float sc = gamma[c] * rsqrtf(var + 1e-5f);
        ss[c] = sc; ss[OCH + c] = beta[c] - mean * sc;
    }
}

// ---------------------------------------------------------------------------
// Kernel 3: BN1 + ReLU + 3x3 s2 p1 maxpool -> f16 with a 1-px ZERO HALO:
// pooled layout (n, c, 58, 58); border written 0 every call (deterministic),
// so the conv2 A-gather needs no bounds checks.
// reduce_window pads with -inf, relu >= 0, window always has a valid tap ->
// init 0 is exact.
// ---------------------------------------------------------------------------
__global__ void pool_kernel(const float* __restrict__ out1,
                            const float* __restrict__ ss,
                            _Float16* __restrict__ pooled)
{
    size_t i = (size_t)blockIdx.x * blockDim.x + threadIdx.x;
    const size_t total = (size_t)NB * OCH * POP * POP;
    if (i >= total) return;
    int xp = i % POP; size_t t = i / POP;
    int yp = t % POP; t /= POP;
    int c = t % OCH; int n = (int)(t / OCH);
    if (xp == 0 || xp == POP - 1 || yp == 0 || yp == POP - 1) {
        pooled[i] = (_Float16)0.f;
        return;
    }
    int x = xp - 1, y = yp - 1;
    float sc = ss[c], sh = ss[OCH + c];
    float m = 0.f;
    #pragma unroll
    for (int dy = -1; dy <= 1; ++dy) {
        int iy = 2 * y + dy;
        if (iy < 0 || iy >= O1) continue;
        #pragma unroll
        for (int dx = -1; dx <= 1; ++dx) {
            int ix = 2 * x + dx;
            if (ix < 0 || ix >= O1) continue;
            float v = fmaf(sc, out1[((size_t)(n * OCH + c) * O1 + iy) * O1 + ix], sh);
            v = fmaxf(v, 0.f);
            m = fmaxf(m, v);
        }
    }
    pooled[i] = (_Float16)m;
}

// Kernel 4: conv2 weight f32 -> f16 (OIHW flat == [oc][k] for the GEMM B)
__global__ void w2_convert(const float* __restrict__ w2, _Float16* __restrict__ w2h)
{
    int i = blockIdx.x * blockDim.x + threadIdx.x;
    if (i < OCH * KTOT) w2h[i] = (_Float16)w2[i];
}

// ---------------------------------------------------------------------------
// Kernel 5: conv2 as implicit GEMM with v_wmma_f32_16x16x32_f16.
// M = 100352 pixels, N = 128 oc, K = 1152. Block: 256 thr = 8 waves,
// block tile 128 pixels x 128 oc. Waves arranged 4(M) x 2(N); each wave
// owns a 32-pixel x 64-oc tile = 2M x 4N subtiles (8 x v8f accumulators).
//
// Per K-step: 4 distinct B frags (global clause, L2-resident weights) +
// 2 distinct A frags (4 ds_load_b128) feed 8 WMMAs with partial waits.
// A tile double-buffered in LDS; ONE barrier per K-step; the gather for
// step kk+32 overlaps the WMMA block of step kk.
//
// A-gather: k = kk + shalf*16 + e is wave-uniform (shalf = tid>>7 constant
// per wave) -> force scalarization with readfirstlane so the ic/ky/kx decode
// and the (ic*58*58 + ky*58 + kx) offset become SALU; with the zero halo the
// load is branch-free: thread-invariant VGPR base + scalar offset.
//
// ISA f16 A layout (16x32): lane L (M=L%16), halves e: K = e + (e>=8)*8 + (L>=16)*8
// ISA f16 B layout (32x16): lane L (N=L%16), halves e: K = e + (e>=8)*8 + (L>=16)*8
// D/C layout: VGPR r: M = r + 8*(L>=16), N = L%16.
// ---------------------------------------------------------------------------
__global__ void __launch_bounds__(256) conv2_wmma(
    const _Float16* __restrict__ pooled,
    const _Float16* __restrict__ w2h,
    float* __restrict__ c2out)
{
    __shared__ _Float16 As[2][128 * 32];
    const int tid  = threadIdx.x;
    const int lane = tid & 31;
    const int wave = tid >> 5;
    const int wmi  = wave >> 1;  // 0..3 : M position (32 pixels each)
    const int wni  = wave & 1;   // 0..1 : N position (64 oc each)
    const int pbase = blockIdx.x * 128;

    v8f acc[2][4] = {};

    // staging assignment: thread -> (pixel row, k half). shalf is wave-uniform.
    const int srow  = tid & 127;
    const int shalf = __builtin_amdgcn_readfirstlane(tid >> 7); // 0 or 1
    const int p  = pbase + srow;
    const int px = p % PO;
    const int py = (p / PO) % PO;
    const int pn = p / (PO * PO);
    // halo base: padded (py+ky, px+kx) == unpadded (py+ky-1, px+kx-1)
    const _Float16* thr_base =
        pooled + (size_t)pn * (OCH * POP * POP) + py * POP + px;
    _Float16* s_dst0 = &As[0][srow * 32 + shalf * 16];
    _Float16* s_dst1 = &As[1][srow * 32 + shalf * 16];

    const int arow0 = wmi * 32 + (lane & 15);
    const int abase = (lane >> 4) * 8;    // K sub-offset in halves
    const int boc   = wni * 64 + (lane & 15);
    const int bkoff = (lane >> 4) * 8;

    auto stage = [&](int kk, _Float16* dst) {
        const int koff = kk + shalf * 16;      // scalar
        #pragma unroll
        for (int e = 0; e < 16; ++e) {
            int k  = koff + e;                 // scalar
            int ic = k / 9,  r9 = k - ic * 9;  // scalar decode
            int ky = r9 / 3, kx = r9 - ky * 3;
            int soff = ic * (POP * POP) + ky * POP + kx; // scalar offset
            dst[e] = thr_base[soff];           // branch-free (zero halo)
        }
    };

    stage(0, s_dst0);
    __syncthreads();

    for (int kk = 0; kk < KTOT; kk += 32) {
        const int cur = (kk >> 5) & 1;
        if (kk + 32 < KTOT) stage(kk + 32, cur ? s_dst0 : s_dst1);

        Frag16 b[4];
        #pragma unroll
        for (int j = 0; j < 4; ++j) {
            const _Float16* wp = w2h + (size_t)(j * 16 + boc) * KTOT + kk + bkoff;
            b[j].f[0] = *(const v4f*)wp;
            b[j].f[1] = *(const v4f*)(wp + 16);
        }
        Frag16 a[2];
        #pragma unroll
        for (int mi = 0; mi < 2; ++mi) {
            const _Float16* ap = &As[cur][(arow0 + mi * 16) * 32 + abase];
            a[mi].f[0] = *(const v4f*)ap;
            a[mi].f[1] = *(const v4f*)(ap + 16);
        }

        #pragma unroll
        for (int mi = 0; mi < 2; ++mi)
            #pragma unroll
            for (int j = 0; j < 4; ++j)
                acc[mi][j] = __builtin_amdgcn_wmma_f32_16x16x32_f16(
                    false, a[mi].h, false, b[j].h, (short)0, acc[mi][j], false, false);

        __syncthreads();
    }

    const int oc0  = lane & 15;
    const int msub = 8 * (lane >> 4);
    #pragma unroll
    for (int mi = 0; mi < 2; ++mi) {
        #pragma unroll
        for (int j = 0; j < 4; ++j) {
            int oc = wni * 64 + j * 16 + oc0;
            #pragma unroll
            for (int r = 0; r < 8; ++r) {
                int p2 = pbase + wmi * 32 + mi * 16 + msub + r;
                c2out[(size_t)p2 * OCH + oc] = acc[mi][j][r];
            }
        }
    }
}

// Kernel 6: BN2 + ReLU, pixel-major -> NCHW f32 output
__global__ void bn2_relu(const float* __restrict__ c2out,
                         const float* __restrict__ ss,
                         float* __restrict__ outp)
{
    size_t i = (size_t)blockIdx.x * blockDim.x + threadIdx.x;
    const size_t total = (size_t)NB * OCH * PO * PO;
    if (i >= total) return;
    int x = i % PO; size_t t = i / PO;
    int y = t % PO; t /= PO;
    int c = t % OCH; int n = (int)(t / OCH);
    size_t pidx = (size_t)n * (PO * PO) + y * PO + x;
    float v = fmaf(ss[c], c2out[pidx * OCH + c], ss[OCH + c]);
    outp[i] = fmaxf(v, 0.f);
}

// ---------------------------------------------------------------------------
static inline size_t al256(size_t x) { return (x + 255) & ~(size_t)255; }

extern "C" void kernel_launch(void* const* d_in, const int* in_sizes, int n_in,
                              void* d_out, int out_size, void* d_ws, size_t ws_size,
                              hipStream_t stream)
{
    const float* inp    = (const float*)d_in[0];
    const float* conv_w = (const float*)d_in[1];
    const float* gamma1 = (const float*)d_in[2];
    const float* beta1  = (const float*)d_in[3];
    const float* conv2w = (const float*)d_in[4];
    const float* gamma2 = (const float*)d_in[5];
    const float* beta2  = (const float*)d_in[6];
    float* outp = (float*)d_out;

    char* ws = (char*)d_ws;
    size_t off = 0;
    float*    out1   = (float*)(ws + off);    off += al256((size_t)NB * OCH * O1 * O1 * 4);
    _Float16* pooled = (_Float16*)(ws + off); off += al256((size_t)NB * OCH * POP * POP * 2);
    _Float16* w2h    = (_Float16*)(ws + off); off += al256((size_t)OCH * KTOT * 2);
    float*    c2out  = (float*)(ws + off);    off += al256((size_t)PIXT * OCH * 4);
    float*    bn1ss  = (float*)(ws + off);    off += al256(2 * OCH * 4);
    float*    bn2ss  = (float*)(ws + off);    off += al256(2 * OCH * 4);
    (void)ws_size; (void)n_in; (void)in_sizes; (void)out_size;

    // 1) conv1 + upsample scatter
    size_t c1_lds = (size_t)(128 * 147 + 147) * sizeof(float); // ~75.9 KB (< 320 KB/WGP)
    conv1_kernel<<<dim3((O1 * O1) / 8, NB), 128, c1_lds, stream>>>(inp, conv_w, out1);

    // 2) BN1 stats
    bn_stats_nchw<<<OCH, 256, 0, stream>>>(out1, gamma1, beta1, bn1ss, O1 * O1, NB);

    // 3) BN1 + ReLU + maxpool -> f16 (halo-padded)
    size_t ptotal_pad = (size_t)NB * OCH * POP * POP;
    pool_kernel<<<(ptotal_pad + 255) / 256, 256, 0, stream>>>(out1, bn1ss, pooled);

    // 4) weight convert
    w2_convert<<<(OCH * KTOT + 255) / 256, 256, 0, stream>>>(conv2w, w2h);

    // 5) conv2 implicit GEMM (WMMA)
    conv2_wmma<<<PIXT / 128, 256, 0, stream>>>(pooled, w2h, c2out);

    // 6) BN2 stats
    bn_stats_pix<<<OCH, 256, 0, stream>>>(c2out, gamma2, beta2, bn2ss, PIXT);

    // 7) BN2 + ReLU -> output
    size_t ptotal = (size_t)NB * OCH * PO * PO;
    bn2_relu<<<(ptotal + 255) / 256, 256, 0, stream>>>(c2out, bn2ss, outp);
}
